// MultiResBlock_27178553049147
// MI455X (gfx1250) — compile-verified, matrix operand order checked by reading, not run on hardware
//
#include <hip/hip_runtime.h>
#include <hip/hip_bf16.h>
#include <math.h>

// ---------------------------------------------------------------------------
// MultiResBlock fused for gfx1250 (MI455X, wave32)
//   B=4, L=8192, D=256, K=2, J=13 levels, N=2D=512
// k0: rmsnorm row reduce -> inv_rms
// k1: conv_w (256x512 f32) -> wT (512x256 f16, transposed)
// k2: 13-level dilated depthwise tree fully LDS-resident (256KB ping-pong),
//     y accumulated in VGPRs, gelu, emit gy as f16
// k3: WMMA f16 GEMM (32768x256 @ 256x512) with double-buffered async
//     global->LDS staging + bias + GLU + skip, f32 out
// ---------------------------------------------------------------------------

#define BATCH   4
#define SEQ     8192
#define DCH     256
#define NLEV    13
#define ROWS    (BATCH * SEQ)          // 32768
#define NOUT    512

typedef __attribute__((ext_vector_type(16))) _Float16 v16h;
typedef __attribute__((ext_vector_type(8)))  _Float16 v8h;
typedef __attribute__((ext_vector_type(8)))  float    v8f;

#if __has_builtin(__builtin_amdgcn_global_load_async_to_lds_b128) && \
    __has_builtin(__builtin_amdgcn_s_wait_asynccnt)
#define USE_ASYNC 1
#else
#define USE_ASYNC 0
#endif

#if USE_ASYNC
// Builtin expects int4 pointers in explicit address spaces
// (AS1 prints as "__device__", AS3 as "__shared__" in HIP diagnostics).
typedef int v4i __attribute__((ext_vector_type(4)));
typedef __attribute__((address_space(1))) v4i glob_v4i;
typedef __attribute__((address_space(3))) v4i lds_v4i;
#endif

// ---------------------------------------------------------------- kernel 0
__global__ __launch_bounds__(256) void rms_kernel(const float* __restrict__ x,
                                                  float* __restrict__ inv_rms) {
    const int lane = threadIdx.x & 31;
    const int row  = blockIdx.x * 8 + (threadIdx.x >> 5);   // 8 rows / block
    const float* p = x + (size_t)row * DCH;
    float s = 0.f;
#pragma unroll
    for (int e = 0; e < 8; ++e) {
        float v = p[lane + 32 * e];
        s += v * v;
    }
#pragma unroll
    for (int o = 16; o > 0; o >>= 1) s += __shfl_xor(s, o, 32);
    if (lane == 0) inv_rms[row] = rsqrtf(s * (1.0f / DCH) + 1e-6f);
}

// ---------------------------------------------------------------- kernel 1
__global__ __launch_bounds__(256) void prep_w_kernel(const float* __restrict__ conv_w,
                                                     _Float16* __restrict__ wT) {
    const int t = blockIdx.x * 256 + threadIdx.x;   // 512*256 total
    const int n = t >> 8;
    const int k = t & 255;
    wT[(size_t)n * DCH + k] = (_Float16)conv_w[(size_t)k * NOUT + n];
}

// ---------------------------------------------------------------- kernel 2
__device__ __forceinline__ float gelu_tanh(float v) {
    // jax.nn.gelu approximate=True
    float c = 0.7978845608028654f * (v + 0.044715f * v * v * v);
    return 0.5f * v * (1.0f + tanhf(c));
}

__global__ __launch_bounds__(1024) void multires_kernel(
        const float* __restrict__ x, const float* __restrict__ inv_rms,
        const float* __restrict__ rms_scale,
        const float* __restrict__ h0, const float* __restrict__ h1,
        const float* __restrict__ w, _Float16* __restrict__ gy) {
    extern __shared__ float lds[];                  // 2 * 4 * 8192 floats
    const int b   = blockIdx.x >> 6;                // 4 batches
    const int c0  = (blockIdx.x & 63) * 4;          // 64 channel groups of 4
    const int t   = threadIdx.x;
    const int c   = t >> 8;                         // local channel 0..3
    const int lb  = t & 255;                        // l base
    const int cg  = c0 + c;

    float* A0 = lds;
    float* A1 = lds + 4 * SEQ;

    const float s0 = h0[cg * 2 + 0], s1 = h0[cg * 2 + 1];
    const float t0 = h1[cg * 2 + 0], t1 = h1[cg * 2 + 1];
    const float w0 = w[cg];
    const float sc = rms_scale[cg];

    float y[32];
    // load + rmsnorm + y init (y = w[0]*xnorm)
#pragma unroll
    for (int i = 0; i < 32; ++i) {
        int l = lb + 256 * i;
        float v = x[((size_t)(b * SEQ + l)) * DCH + cg] * inv_rms[b * SEQ + l] * sc;
        A0[c * SEQ + l] = v;
        y[i] = w0 * v;
    }
    __syncthreads();

    float* cur = A0;
    float* nxt = A1;
    for (int j = 0; j < NLEV; ++j) {
        const int   dil = 1 << j;
        const float wj  = w[(j + 1) * DCH + cg];
#pragma unroll
        for (int i = 0; i < 32; ++i) {
            int   l  = lb + 256 * i;
            float av = cur[c * SEQ + l];
            float am = (l >= dil) ? cur[c * SEQ + l - dil] : 0.0f;
            y[i] += wj * (am * t0 + av * t1);       // b-branch (h1)
            nxt[c * SEQ + l] = am * s0 + av * s1;   // a-branch (h0)
        }
        __syncthreads();
        float* tmp = cur; cur = nxt; nxt = tmp;
    }

    const float wl = w[(NLEV + 1) * DCH + cg];
#pragma unroll
    for (int i = 0; i < 32; ++i) {
        int   l = lb + 256 * i;
        float v = y[i] + wl * cur[c * SEQ + l];
        gy[((size_t)(b * SEQ + l)) * DCH + cg] = (_Float16)gelu_tanh(v);
    }
}

// ---------------------------------------------------------------- kernel 3
#define LSTR 40   // padded LDS row stride in halves (80B, 16B aligned)

__device__ __forceinline__ v16h frag16(const _Float16* p) {
    // ISA 16-bit fragment: elements 0..7 = K[8h..8h+7], 8..15 = K[16+8h..16+8h+7]
    v8h lo = *(const v8h*)(p);
    v8h hi = *(const v8h*)(p + 16);
    v16h r;
#pragma unroll
    for (int e = 0; e < 8; ++e) { r[e] = lo[e]; r[e + 8] = hi[e]; }
    return r;
}

#if USE_ASYNC
__device__ __forceinline__ void async_ld16(const _Float16* g, _Float16* l) {
    // GLOBAL_LOAD_ASYNC_TO_LDS_B128: 16B per lane, ASYNCcnt-tracked, no VGPR data
    __builtin_amdgcn_global_load_async_to_lds_b128(
        (glob_v4i*)g, (lds_v4i*)l, 0, 0);
}
#endif

__global__ __launch_bounds__(256) void gemm_glu_kernel(
        const _Float16* __restrict__ gy, const _Float16* __restrict__ wT,
        const float* __restrict__ conv_b, const float* __restrict__ x,
        float* __restrict__ out) {
    __shared__ __align__(16) _Float16 Al[2][128 * LSTR];
    __shared__ __align__(16) _Float16 Bv[2][64 * LSTR];
    __shared__ __align__(16) _Float16 Bg[2][64 * LSTR];

    const int m0   = blockIdx.x * 128;
    const int n0   = blockIdx.y * 64;
    const int t    = threadIdx.x;
    const int lane = t & 31;
    const int wid  = t >> 5;
    const int wm   = wid >> 1;      // 0..3 -> 32-row slice
    const int wn   = wid & 1;       // 0..1 -> 32-col slice
    const int kh   = lane >> 4;     // K-half select
    const int r15  = lane & 15;

    const int rowA = t >> 1, offA = (t & 1) * 16;   // A: 128 rows x 32 halves
    const int rowB = t >> 2, offB = (t & 3) * 8;    // B: 64 rows x 32 halves (x2)

    auto stage = [&](int k0, int buf) {
#if USE_ASYNC
        async_ld16(gy + (size_t)(m0 + rowA) * DCH + k0 + offA,
                   &Al[buf][rowA * LSTR + offA]);
        async_ld16(gy + (size_t)(m0 + rowA) * DCH + k0 + offA + 8,
                   &Al[buf][rowA * LSTR + offA + 8]);
        async_ld16(wT + (size_t)(n0 + rowB) * DCH + k0 + offB,
                   &Bv[buf][rowB * LSTR + offB]);
        async_ld16(wT + (size_t)(n0 + 256 + rowB) * DCH + k0 + offB,
                   &Bg[buf][rowB * LSTR + offB]);
#else
        const uint4* s = (const uint4*)(gy + (size_t)(m0 + rowA) * DCH + k0 + offA);
        *(uint4*)&Al[buf][rowA * LSTR + offA]     = s[0];
        *(uint4*)&Al[buf][rowA * LSTR + offA + 8] = s[1];
        *(uint4*)&Bv[buf][rowB * LSTR + offB] =
            *(const uint4*)(wT + (size_t)(n0 + rowB) * DCH + k0 + offB);
        *(uint4*)&Bg[buf][rowB * LSTR + offB] =
            *(const uint4*)(wT + (size_t)(n0 + 256 + rowB) * DCH + k0 + offB);
#endif
    };

    v8f accv[2][2] = {};
    v8f accg[2][2] = {};

    stage(0, 0);                     // prologue: tile 0 in flight
#pragma unroll
    for (int s8 = 0; s8 < 8; ++s8) { // K = 256 in steps of 32
        const int cb = s8 & 1;
        if (s8 < 7) {
            stage((s8 + 1) * 32, cb ^ 1);   // overlap next tile with compute
#if USE_ASYNC
            __builtin_amdgcn_s_wait_asynccnt(4);  // in-order: tile s8 resident
#endif
        } else {
#if USE_ASYNC
            __builtin_amdgcn_s_wait_asynccnt(0);
#endif
        }
        __syncthreads();

        v16h af[2], bvf[2], bgf[2];
#pragma unroll
        for (int i = 0; i < 2; ++i)
            af[i] = frag16(&Al[cb][(wm * 32 + 16 * i + r15) * LSTR + 8 * kh]);
#pragma unroll
        for (int j = 0; j < 2; ++j) {
            bvf[j] = frag16(&Bv[cb][(wn * 32 + 16 * j + r15) * LSTR + 8 * kh]);
            bgf[j] = frag16(&Bg[cb][(wn * 32 + 16 * j + r15) * LSTR + 8 * kh]);
        }
#pragma unroll
        for (int i = 0; i < 2; ++i)
#pragma unroll
            for (int j = 0; j < 2; ++j) {
                accv[i][j] = __builtin_amdgcn_wmma_f32_16x16x32_f16(
                    false, af[i], false, bvf[j], (short)0, accv[i][j], false, false);
                accg[i][j] = __builtin_amdgcn_wmma_f32_16x16x32_f16(
                    false, af[i], false, bgf[j], (short)0, accg[i][j], false, false);
            }
        __syncthreads();   // all waves done reading buf cb before it is re-staged
    }

    // epilogue: bias + GLU + skip   (C/D layout: M = r + 8*kh, N = lane&15)
#pragma unroll
    for (int j = 0; j < 2; ++j) {
        const int   n  = n0 + wn * 32 + 16 * j + r15;
        const float bv = conv_b[n];
        const float bg = conv_b[n + 256];
#pragma unroll
        for (int i = 0; i < 2; ++i) {
            const int mb = m0 + wm * 32 + 16 * i + 8 * kh;
#pragma unroll
            for (int r = 0; r < 8; ++r) {
                const int   m    = mb + r;
                const float val  = accv[i][j][r] + bv;
                const float gate = accg[i][j][r] + bg;
                const float g    = val / (1.0f + expf(-gate));
                out[(size_t)m * DCH + n] = g + x[(size_t)m * DCH + n];
            }
        }
    }
}

// ---------------------------------------------------------------- launcher
extern "C" void kernel_launch(void* const* d_in, const int* in_sizes, int n_in,
                              void* d_out, int out_size, void* d_ws, size_t ws_size,
                              hipStream_t stream) {
    const float* x         = (const float*)d_in[0];   // (B,L,D)
    const float* rms_scale = (const float*)d_in[1];   // (D)
    const float* h0        = (const float*)d_in[2];   // (D,2)
    const float* h1        = (const float*)d_in[3];   // (D,2)
    const float* w         = (const float*)d_in[4];   // (15,D)
    const float* conv_w    = (const float*)d_in[5];   // (D,2D)
    const float* conv_b    = (const float*)d_in[6];   // (2D)
    float* out             = (float*)d_out;           // (B,L,D)

    // workspace layout
    float*    inv_rms = (float*)d_ws;                                       // 128 KB
    _Float16* wT      = (_Float16*)((char*)d_ws + 131072);                  // 256 KB
    _Float16* gy      = (_Float16*)((char*)d_ws + 131072 + 262144);         // 16 MB

    const int ldsBytes = 2 * 4 * SEQ * (int)sizeof(float);                  // 256 KB
    (void)hipFuncSetAttribute((const void*)multires_kernel,
                              hipFuncAttributeMaxDynamicSharedMemorySize, ldsBytes);

    rms_kernel<<<ROWS / 8, 256, 0, stream>>>(x, inv_rms);
    prep_w_kernel<<<(NOUT * DCH) / 256, 256, 0, stream>>>(conv_w, wT);
    multires_kernel<<<BATCH * (DCH / 4), 1024, ldsBytes, stream>>>(
        x, inv_rms, rms_scale, h0, h1, w, gy);
    gemm_glu_kernel<<<dim3(ROWS / 128, DCH / 64), 256, 0, stream>>>(
        gy, wT, conv_b, x, out);

    (void)in_sizes; (void)n_in; (void)out_size; (void)ws_size;
}
